// BlockSoftmaxLinearHybrid_8650064134243
// MI455X (gfx1250) — matrix operand order repaired and checked by reading, not compiled, not run on hardware
//
#include <hip/hip_runtime.h>
#include <stdint.h>

// ---- problem constants (from reference setup_inputs) ----
constexpr int Bb = 2;          // batch
constexpr int Hh = 32;         // heads
constexpr int Ll = 4096;       // sequence length
constexpr int Dd = 64;         // head dim
constexpr int Ff = 64;         // hedgehog feature dim (phi width = 2F)
constexpr int Ss = 32;         // block size
constexpr int Nn = Ll / Ss;    // 128 blocks (sequential scan length)
constexpr int DF = 2 * Ff;     // 128

// LDS row strides in halves; multiples of 8 so every fragment load is 16B aligned
constexpr int LDW   = Dd + 8;  // 72   W^T rows (len D)
constexpr int LDQ   = Dd + 8;  // 72   q/k rows
constexpr int LDVT  = Ss + 8;  // 40   v^T rows (len S)
constexpr int LDPQ  = DF + 8;  // 136  phi_q rows (len 2F)
constexpr int LDPKT = Ss + 8;  // 40   phi_k^T rows (len S)
constexpr int LDATT = Ss + 8;  // 40   attn rows (len S)

typedef _Float16 v16h __attribute__((ext_vector_type(16)));
typedef _Float16 v8h  __attribute__((ext_vector_type(8)));
typedef float    v8f  __attribute__((ext_vector_type(8)));
typedef float    v4f  __attribute__((ext_vector_type(4)));

#define DEV static __device__ __forceinline__

// D(16x16 f32) = A(16x32 f16) * B(32x16 f16) + C  -- v_wmma_f32_16x16x32_f16
DEV v8f wmma16(v16h a, v16h b, v8f c) {
  return __builtin_amdgcn_wmma_f32_16x16x32_f16(false, a, false, b,
                                                (short)0, c, false, false);
}

// Per-lane async DMA: LDS[ldsoff] = MEM[gbase + goff] (16 bytes), ASYNCcnt++
DEV void async_copy_b128(uint32_t ldsoff, uint64_t gbase, uint32_t goff) {
  asm volatile("global_load_async_to_lds_b128 %0, %1, %2"
               :: "v"(ldsoff), "v"(goff), "s"(gbase) : "memory");
}
DEV void wait_asynccnt0() {
  asm volatile("s_wait_asynccnt 0x0" ::: "memory");
}

// A fragment (16x32) from row-major f16 LDS. lanes0-15 M=lane,
// halves[0:15]=K+0..15; lanes16-31 shift K by +16.
DEV v16h load_a(const _Float16* base, int ld, int m0, int k0, int lane) {
  const int m  = m0 + (lane & 15);
  const int kk = k0 + ((lane & 16) ? 16 : 0);
  const _Float16* p = base + m * ld + kk;
  v8h lo = *(const v8h*)(p);
  v8h hi = *(const v8h*)(p + 8);
  v16h a;
#pragma unroll
  for (int i = 0; i < 8; ++i) { a[i] = lo[i]; a[i + 8] = hi[i]; }
  return a;
}

// B fragment (32x16) from N-major (transposed) f16 staging bt[n][k]:
// lane holds column n = n0+(lane&15), 16 consecutive K (lanes16-31: K+16).
DEV v16h load_b(const _Float16* bt, int ld, int n0, int k0, int lane) {
  const int n  = n0 + (lane & 15);
  const int kk = k0 + ((lane & 16) ? 16 : 0);
  const _Float16* p = bt + n * ld + kk;
  v8h lo = *(const v8h*)(p);
  v8h hi = *(const v8h*)(p + 8);
  v16h b;
#pragma unroll
  for (int i = 0; i < 8; ++i) { b[i] = lo[i]; b[i + 8] = hi[i]; }
  return b;
}

// B fragment from N-major f32 storage (running state), converted to f16.
DEV v16h load_b_f32(const float* bt, int ld, int n0, int k0, int lane) {
  const int n  = n0 + (lane & 15);
  const int kk = k0 + ((lane & 16) ? 16 : 0);
  const float* p = bt + n * ld + kk;
  v4f x0 = *(const v4f*)(p);
  v4f x1 = *(const v4f*)(p + 4);
  v4f x2 = *(const v4f*)(p + 8);
  v4f x3 = *(const v4f*)(p + 12);
  v16h b;
#pragma unroll
  for (int i = 0; i < 4; ++i) {
    b[i]      = (_Float16)x0[i];
    b[i + 4]  = (_Float16)x1[i];
    b[i + 8]  = (_Float16)x2[i];
    b[i + 12] = (_Float16)x3[i];
  }
  return b;
}

// Store C/D tile to row-major c[m][n]. ISA: VGPR r = M=r (lanes0-15) / M=r+8.
DEV void store_c_rm(float* c, int ld, int m0, int n0, v8f acc, int lane) {
  const int n  = n0 + (lane & 15);
  const int mb = m0 + ((lane & 16) ? 8 : 0);
#pragma unroll
  for (int r = 0; r < 8; ++r) c[(mb + r) * ld + n] = acc[r];
}

// Load/store C/D tile from N-major storage st[n][m]: 8 consecutive f32 per lane.
DEV v8f load_c_nm(const float* st, int ld, int m0, int n0, int lane) {
  const int n  = n0 + (lane & 15);
  const int mb = m0 + ((lane & 16) ? 8 : 0);
  const float* p = st + n * ld + mb;
  v4f a = *(const v4f*)(p);
  v4f b = *(const v4f*)(p + 4);
  v8f c;
#pragma unroll
  for (int i = 0; i < 4; ++i) { c[i] = a[i]; c[i + 4] = b[i]; }
  return c;
}
DEV void store_c_nm(float* st, int ld, int m0, int n0, v8f c, int lane) {
  const int n  = n0 + (lane & 15);
  const int mb = m0 + ((lane & 16) ? 8 : 0);
  float* p = st + n * ld + mb;
  v4f a, b;
#pragma unroll
  for (int i = 0; i < 4; ++i) { a[i] = c[i]; b[i] = c[i + 4]; }
  *(v4f*)(p)     = a;
  *(v4f*)(p + 4) = b;
}

// ---- LDS budget (dynamic) ----
constexpr size_t H_ELEMS = (size_t)Ff * LDW + 2u * Ss * LDQ + (size_t)Dd * LDVT +
                           (size_t)Ss * LDPQ + (size_t)DF * LDPKT + (size_t)Ss * LDATT;
constexpr size_t F_ELEMS = 2u * Ss * Ff + (size_t)Ss * Ss + 2u * Ss * Dd +
                           (size_t)Dd * DF + DF + Ss;
constexpr uint32_t STAGE_OFF   = (uint32_t)(H_ELEMS * 2 + F_ELEMS * 4);  // 115,328
constexpr uint32_t STAGE_BYTES = 3u * Ss * Dd * 4u;                      // 24,576 / buffer
constexpr size_t   SMEM_BYTES  = STAGE_OFF + 2u * STAGE_BYTES;           // 164,480 < 320KB

__global__ __launch_bounds__(256)
void hedgehog_hybrid_kernel(const float* __restrict__ Q, const float* __restrict__ K,
                            const float* __restrict__ V, const float* __restrict__ W,
                            const float* __restrict__ alpha, float* __restrict__ Out) {
  extern __shared__ char smem[];
  _Float16* sW   = (_Float16*)smem;            // W^T  [f][d]
  _Float16* sQ   = sW  + Ff * LDW;             // q    [s][d]
  _Float16* sK   = sQ  + Ss * LDQ;             // k    [t][d]
  _Float16* sVT  = sK  + Ss * LDQ;             // v^T  [d][s]
  _Float16* sPQ  = sVT + Dd * LDVT;            // phi_q [s][2F]
  _Float16* sPKT = sPQ + Ss * LDPQ;            // phi_k^T [2F][s]
  _Float16* sATT = sPKT + DF * LDPKT;          // attn [s][t]
  float* fb   = (float*)(smem + H_ELEMS * 2);
  float* sUQ  = fb;                            // u_q  [s][F]
  float* sUK  = sUQ  + Ss * Ff;                // u_k  [s][F]
  float* sSC  = sUK  + Ss * Ff;                // scores [s][t]
  float* sSM  = sSC  + Ss * Ss;                // softmax branch out [s][d]
  float* sLIN = sSM  + Ss * Dd;                // linear branch out  [s][d]
  float* sST  = sLIN + Ss * Dd;                // state S^T [d][2F]  (f32)
  float* sZ   = sST  + Dd * DF;                // state Z  [2F]
  float* sDEN = sZ   + DF;                     // denom    [s]
  const float* sStage = (const float*)(smem + STAGE_OFF);  // 2 x [q|k|v] f32 staging

  const int tid  = threadIdx.x;
  const int wv   = tid >> 5;
  const int lane = tid & 31;
  const int r8   = tid >> 3;                   // row id for 8-lane-per-row phases
  const int sub  = tid & 7;                    // lane-in-row
  const int bh   = blockIdx.x;                 // (b*H + h)
  const int h    = bh % Hh;

  const float* qg = Q   + (size_t)bh * Ll * Dd;
  const float* kg = K   + (size_t)bh * Ll * Dd;
  const float* vg = V   + (size_t)bh * Ll * Dd;
  float*       og = Out + (size_t)bh * Ll * Dd;
  const float* wg = W   + (size_t)h * Dd * Ff;

  // LDS byte offsets for async DMA (dynamic LDS starts after static LDS)
  const uint32_t dynbase = __builtin_amdgcn_groupstaticsize();
  const uint64_t qg64 = (uint64_t)(uintptr_t)qg;
  const uint64_t kg64 = (uint64_t)(uintptr_t)kg;
  const uint64_t vg64 = (uint64_t)(uintptr_t)vg;

  const float wsig = 1.0f / (1.0f + __expf(-alpha[0]));
  const float wlin = 1.0f - wsig;

  // Stage per-head weights transposed: sW[f][d] = W[h][d][f]
  for (int i = tid; i < Dd * Ff; i += 256) {
    int d = i / Ff, f = i % Ff;
    sW[f * LDW + d] = (_Float16)wg[i];
  }
  for (int i = tid; i < Dd * DF; i += 256) sST[i] = 0.0f;   // S = 0
  if (tid < DF) sZ[tid] = 0.0f;                             // Z = 0

  // Prologue: kick off async copy of block 0 into staging buffer 0
  {
    const uint32_t base = dynbase + STAGE_OFF;
#pragma unroll
    for (int j = 0; j < 2; ++j) {
      const uint32_t e = (uint32_t)(tid + j * 256) * 16u;   // byte offset of b128
      async_copy_b128(base + e,          qg64, e);
      async_copy_b128(base + 8192u + e,  kg64, e);
      async_copy_b128(base + 16384u + e, vg64, e);
    }
  }
  __syncthreads();

  for (int n = 0; n < Nn; ++n) {
    const int cur = n & 1;

    // Retire this block's async DMA (per-wave ASYNCcnt), make LDS visible to WG
    wait_asynccnt0();
    __syncthreads();

    // Issue async DMA for the NEXT block into the other staging buffer;
    // its latency is hidden behind this entire block-step.
    if (n + 1 < Nn) {
      const uint32_t base = dynbase + STAGE_OFF + (uint32_t)(cur ^ 1) * STAGE_BYTES;
      const uint32_t goff = (uint32_t)((n + 1) * Ss * Dd * 4);
#pragma unroll
      for (int j = 0; j < 2; ++j) {
        const uint32_t e = (uint32_t)(tid + j * 256) * 16u;
        async_copy_b128(base + e,          qg64, goff + e);
        async_copy_b128(base + 8192u + e,  kg64, goff + e);
        async_copy_b128(base + 16384u + e, vg64, goff + e);
      }
    }

    // Convert staged f32 -> f16 operand tiles (q,k row-major; v transposed)
    {
      const float* stq = sStage + (size_t)cur * (3 * Ss * Dd);
      const float* stk = stq + Ss * Dd;
      const float* stv = stk + Ss * Dd;
      for (int i = tid; i < (Ss * Dd) / 4; i += 256) {
        int e = i * 4;
        int s = e >> 6, d = e & 63;
        v4f xq = *(const v4f*)(stq + e);
        v4f xk = *(const v4f*)(stk + e);
        v4f xv = *(const v4f*)(stv + e);
#pragma unroll
        for (int j = 0; j < 4; ++j) {
          sQ[s * LDQ + d + j]     = (_Float16)xq[j];
          sK[s * LDQ + d + j]     = (_Float16)xk[j];
          sVT[(d + j) * LDVT + s] = (_Float16)xv[j];
        }
      }
    }
    __syncthreads();

    // Phase 1 (WMMA): u_q = q@W, u_k = k@W (8 tiles each, wave=tile),
    //                 scores = q@k^T * D^-1/2 (4 tiles, waves 0-3)
    {
      const int mt = (wv >> 2) & 1, nt = wv & 3;
      v8f accq = {}; v8f acck = {};
#pragma unroll
      for (int kk = 0; kk < Dd; kk += 32) {
        v16h bw = load_b(sW, LDW, nt * 16, kk, lane);
        accq = wmma16(load_a(sQ, LDQ, mt * 16, kk, lane), bw, accq);
        acck = wmma16(load_a(sK, LDQ, mt * 16, kk, lane), bw, acck);
      }
      store_c_rm(sUQ, Ff, mt * 16, nt * 16, accq, lane);
      store_c_rm(sUK, Ff, mt * 16, nt * 16, acck, lane);
    }
    if (wv < 4) {
      const int mt = wv >> 1, nt = wv & 1;
      v8f acc = {};
#pragma unroll
      for (int kk = 0; kk < Dd; kk += 32)
        acc = wmma16(load_a(sQ, LDQ, mt * 16, kk, lane),
                     load_b(sK, LDQ, nt * 16, kk, lane), acc);
#pragma unroll
      for (int r = 0; r < 8; ++r) acc[r] *= 0.125f;         // D^-0.5
      store_c_rm(sSC, Ss, mt * 16, nt * 16, acc, lane);
    }
    __syncthreads();

    // Phase 2 (VALU, fully occupied): 8 lanes per row, values in registers,
    // reductions via __shfl_xor within 8-lane groups.
    {   // phi_q row r8: [softmax(u), softmax(-u)] -> sPQ (row-major)
      const float* u = sUQ + r8 * Ff + sub * 8;
      v4f a0 = *(const v4f*)(u);
      v4f a1 = *(const v4f*)(u + 4);
      float uv[8];
#pragma unroll
      for (int j = 0; j < 4; ++j) { uv[j] = a0[j]; uv[4 + j] = a1[j]; }
      float mx = uv[0], mn = uv[0];
#pragma unroll
      for (int j = 1; j < 8; ++j) { mx = fmaxf(mx, uv[j]); mn = fminf(mn, uv[j]); }
#pragma unroll
      for (int m = 1; m < 8; m <<= 1) {
        mx = fmaxf(mx, __shfl_xor(mx, m, 8));
        mn = fminf(mn, __shfl_xor(mn, m, 8));
      }
      float ep[8], em[8], sp = 0.f, sm = 0.f;
#pragma unroll
      for (int j = 0; j < 8; ++j) {
        ep[j] = __expf(uv[j] - mx); sp += ep[j];
        em[j] = __expf(mn - uv[j]); sm += em[j];
      }
#pragma unroll
      for (int m = 1; m < 8; m <<= 1) {
        sp += __shfl_xor(sp, m, 8);
        sm += __shfl_xor(sm, m, 8);
      }
      float rp = 1.f / sp, rm = 1.f / sm;
      v8h hp, hm;
#pragma unroll
      for (int j = 0; j < 8; ++j) {
        hp[j] = (_Float16)(ep[j] * rp);
        hm[j] = (_Float16)(em[j] * rm);
      }
      *(v8h*)(sPQ + r8 * LDPQ + sub * 8)      = hp;
      *(v8h*)(sPQ + r8 * LDPQ + Ff + sub * 8) = hm;
    }
    {   // phi_k row r8 -> sPKT (transposed write for the S-update A-operand)
      const float* u = sUK + r8 * Ff + sub * 8;
      v4f a0 = *(const v4f*)(u);
      v4f a1 = *(const v4f*)(u + 4);
      float uv[8];
#pragma unroll
      for (int j = 0; j < 4; ++j) { uv[j] = a0[j]; uv[4 + j] = a1[j]; }
      float mx = uv[0], mn = uv[0];
#pragma unroll
      for (int j = 1; j < 8; ++j) { mx = fmaxf(mx, uv[j]); mn = fminf(mn, uv[j]); }
#pragma unroll
      for (int m = 1; m < 8; m <<= 1) {
        mx = fmaxf(mx, __shfl_xor(mx, m, 8));
        mn = fminf(mn, __shfl_xor(mn, m, 8));
      }
      float ep[8], em[8], sp = 0.f, sm = 0.f;
#pragma unroll
      for (int j = 0; j < 8; ++j) {
        ep[j] = __expf(uv[j] - mx); sp += ep[j];
        em[j] = __expf(mn - uv[j]); sm += em[j];
      }
#pragma unroll
      for (int m = 1; m < 8; m <<= 1) {
        sp += __shfl_xor(sp, m, 8);
        sm += __shfl_xor(sm, m, 8);
      }
      float rp = 1.f / sp, rm = 1.f / sm;
#pragma unroll
      for (int j = 0; j < 8; ++j) {
        int f = sub * 8 + j;
        sPKT[f * LDPKT + r8]        = (_Float16)(ep[j] * rp);
        sPKT[(Ff + f) * LDPKT + r8] = (_Float16)(em[j] * rm);
      }
    }
    {   // attention softmax, row r8, 4 cols per lane
      const float* sc = sSC + r8 * Ss + sub * 4;
      v4f x = *(const v4f*)(sc);
      float mx = fmaxf(fmaxf(x[0], x[1]), fmaxf(x[2], x[3]));
#pragma unroll
      for (int m = 1; m < 8; m <<= 1) mx = fmaxf(mx, __shfl_xor(mx, m, 8));
      float e0 = __expf(x[0] - mx), e1 = __expf(x[1] - mx);
      float e2 = __expf(x[2] - mx), e3 = __expf(x[3] - mx);
      float sum = e0 + e1 + e2 + e3;
#pragma unroll
      for (int m = 1; m < 8; m <<= 1) sum += __shfl_xor(sum, m, 8);
      float inv = 1.f / sum;
      _Float16* at = sATT + r8 * LDATT + sub * 4;
      at[0] = (_Float16)(e0 * inv); at[1] = (_Float16)(e1 * inv);
      at[2] = (_Float16)(e2 * inv); at[3] = (_Float16)(e3 * inv);
    }
    __syncthreads();

    // Phase 3 (WMMA): sm = attn@v ; lin = phi_q @ S_old (state consumed pre-update)
    {
      const int mt = (wv >> 2) & 1, nt = wv & 3;
      v8f accs = {};
      accs = wmma16(load_a(sATT, LDATT, mt * 16, 0, lane),
                    load_b(sVT, LDVT, nt * 16, 0, lane), accs);
      store_c_rm(sSM, Dd, mt * 16, nt * 16, accs, lane);

      v8f accl = {};
#pragma unroll
      for (int kk = 0; kk < DF; kk += 32)
        accl = wmma16(load_a(sPQ, LDPQ, mt * 16, kk, lane),
                      load_b_f32(sST, DF, nt * 16, kk, lane), accl);
      store_c_rm(sLIN, Dd, mt * 16, nt * 16, accl, lane);
    }
    __syncthreads();

    // Phase 4a: denom = max(phi_q . Z_old, eps); 8 lanes per row
    {
      const _Float16* pq = sPQ + r8 * LDPQ + sub * 16;
      const float*    zz = sZ + sub * 16;
      float den = 0.f;
#pragma unroll
      for (int j = 0; j < 16; ++j) den += (float)pq[j] * zz[j];
#pragma unroll
      for (int m = 1; m < 8; m <<= 1) den += __shfl_xor(den, m, 8);
      if (sub == 0) sDEN[r8] = fmaxf(den, 1.0e-6f);
    }
    __syncthreads();

    // Phase 4b: out = sigmoid(a)*sm + (1-sigmoid(a))*lin/denom
    {
      float* ob = og + n * Ss * Dd;
      for (int i = tid; i < Ss * Dd; i += 256) {
        int s = i >> 6;
        ob[i] = wsig * sSM[i] + wlin * (sLIN[i] / sDEN[s]);
      }
    }
    __syncthreads();

    // Phase 5a (WMMA): S += phi_k^T @ v (f32 state accumulated via C operand)
    {
      const int m0 = wv * 16;                       // feature block (2F = 8 blocks)
      v16h a = load_a(sPKT, LDPKT, m0, 0, lane);
#pragma unroll
      for (int nt = 0; nt < 4; ++nt) {
        const int n0 = nt * 16;                     // d block
        v8f c = load_c_nm(sST, DF, m0, n0, lane);
        c = wmma16(a, load_b(sVT, LDVT, n0, 0, lane), c);
        store_c_nm(sST, DF, m0, n0, c, lane);
      }
    }
    __syncthreads();

    // Phase 5b: Z += colsum(phi_k)
    if (tid < DF) {
      float z = 0.f;
      for (int s = 0; s < Ss; ++s) z += (float)sPKT[tid * LDPKT + s];
      sZ[tid] += z;
    }
    __syncthreads();
  }
}

extern "C" void kernel_launch(void* const* d_in, const int* in_sizes, int n_in,
                              void* d_out, int out_size, void* d_ws, size_t ws_size,
                              hipStream_t stream) {
  const float* q = (const float*)d_in[0];
  const float* k = (const float*)d_in[1];
  const float* v = (const float*)d_in[2];
  const float* w = (const float*)d_in[3];
  const float* a = (const float*)d_in[4];
  float* out = (float*)d_out;
  (void)in_sizes; (void)n_in; (void)out_size; (void)d_ws; (void)ws_size;

  hedgehog_hybrid_kernel<<<dim3(Bb * Hh), dim3(256), SMEM_BYTES, stream>>>(
      q, k, v, w, a, out);
}
